// MultiHeadAttention_40209483825883
// MI455X (gfx1250) — matrix operand order repaired and checked by reading, not compiled, not run on hardware
//
#include <hip/hip_runtime.h>
#include <hip/hip_bf16.h>

typedef _Float16 v4h  __attribute__((ext_vector_type(4)));
typedef _Float16 v8h  __attribute__((ext_vector_type(8)));
typedef _Float16 v16h __attribute__((ext_vector_type(16)));
typedef float    v8f  __attribute__((ext_vector_type(8)));

union V16 { v16h v; v8h h[2]; };

#define D_MODEL 1024
#define SEQ     2048
#define NB      2
#define NH      16
#define DK      64
#define LDKC    72     // attn K-chunk LDS stride: 144B rows, 16B-aligned
#define LDVC    40     // attn V-chunk LDS stride: 80B rows, 16B-aligned
#define LDP     40     // attn P-buffer stride

#if defined(__has_builtin)
#if __has_builtin(__builtin_amdgcn_sched_group_barrier)
#define SCHED_GROUP(mask, size, id) __builtin_amdgcn_sched_group_barrier(mask, size, id)
#endif
#endif
#ifndef SCHED_GROUP
#define SCHED_GROUP(mask, size, id)
#endif
// masks: 0x008 = MFMA/WMMA, 0x020 = VMEM read, 0x100 = DS read

__device__ __forceinline__ v8f wmma16(v16h a, v16h b, v8f c) {
    // D = A(16x32 f16) * B(32x16 f16) + C(16x16 f32)
    return __builtin_amdgcn_wmma_f32_16x16x32_f16(
        false, a, false, b, (short)0, c, false, false);
}

// ---------------------------------------------------------------------------
// Kernel 0: fp32 -> f16 streaming conversion (weights AND activations)
// ---------------------------------------------------------------------------
__global__ __launch_bounds__(256) void cvt16(const float4* __restrict__ src,
                                             v4h* __restrict__ dst, int n4) {
    int i = blockIdx.x * blockDim.x + threadIdx.x;
    if (i < n4) {
        float4 f = src[i];
        v4h h = { (_Float16)f.x, (_Float16)f.y, (_Float16)f.z, (_Float16)f.w };
        dst[i] = h;
    }
}

// ---------------------------------------------------------------------------
// Fragment loader for the 64x16-per-wave GEMM tiles (1 B-frag + 4 A-frags)
// ---------------------------------------------------------------------------
__device__ __forceinline__ void load_frags(const _Float16* __restrict__ ar0,
                                           const _Float16* __restrict__ br,
                                           int kb, int o8, int o16,
                                           V16& b, V16 (&a)[4]) {
    b.h[0] = *(const v8h*)(br + kb + o16);
    b.h[1] = *(const v8h*)(br + kb + o16 + 8);
#pragma unroll
    for (int s = 0; s < 4; ++s) {
        a[s].h[0] = *(const v8h*)(ar0 + s * 16 * D_MODEL + kb + o16);
        a[s].h[1] = *(const v8h*)(ar0 + s * 16 * D_MODEL + kb + o16 + 8);
    }
}

// ---------------------------------------------------------------------------
// Kernel 1: projection GEMM (X16 @ W^T) + RoPE + head-split (+V transpose)
// Block = 4 waves; wave computes 64x16 (4 M-subtiles, B-fragment reused 4x).
// Software-pipelined + sched_group_barrier: [10 VMEM-read][4 WMMA] per iter.
// grid: x = D/16/4 = 16, y = B*S/64 = 64, z = 3
// ---------------------------------------------------------------------------
__global__ __launch_bounds__(128) void proj_rope(
    const _Float16* __restrict__ Xq, const _Float16* __restrict__ Xk,
    const _Float16* __restrict__ Xv,
    const _Float16* __restrict__ Wq, const _Float16* __restrict__ Wk,
    const _Float16* __restrict__ Wv,
    _Float16* __restrict__ Qo, _Float16* __restrict__ Ko,
    _Float16* __restrict__ Vo)
{
    const int z = blockIdx.z;
    const _Float16* X   = (z == 0) ? Xq : (z == 1) ? Xk : Xv;
    const _Float16* W   = (z == 0) ? Wq : (z == 1) ? Wk : Wv;
    _Float16*       dst = (z == 0) ? Qo : (z == 1) ? Ko : Vo;
    const bool vtrans = (z == 2);

    const int tid   = threadIdx.x;
    const int w     = tid >> 5;
    const int lane  = tid & 31;
    const int mbase = blockIdx.y * 64;
    const int nbase = (blockIdx.x * 4 + w) * 16;
    const int mrow  = lane & 15;
    const int o8    = (lane >> 4) * 8;
    const int o16   = (lane >> 4) * 16;
    const int hi8   = (lane >> 4) * 8;

    const _Float16* ar0 = X + (size_t)(mbase + mrow) * D_MODEL;
    const _Float16* br  = W + (size_t)(nbase + mrow) * D_MODEL;

    v8f acc[4] = { {}, {}, {}, {} };

    V16 bc, ac[4];
    load_frags(ar0, br, 0, o8, o16, bc, ac);
    for (int kb = 32; kb < D_MODEL; kb += 32) {
        V16 bn, an[4];
        load_frags(ar0, br, kb, o8, o16, bn, an);
#pragma unroll
        for (int s = 0; s < 4; ++s)
            acc[s] = wmma16(ac[s].v, bc.v, acc[s]);
        // steady-state shape: issue all 10 next-iter loads, then 4 WMMAs
        SCHED_GROUP(0x020, 10, 0);
        SCHED_GROUP(0x008, 4, 0);
        bc = bn;
#pragma unroll
        for (int s = 0; s < 4; ++s) ac[s] = an[s];
    }
#pragma unroll
    for (int s = 0; s < 4; ++s)
        acc[s] = wmma16(ac[s].v, bc.v, acc[s]);

    // RoPE epilogue: channel d fixed per lane; partner channel is lane^1
    const int d = nbase + mrow;
    const int j = d >> 1;
    const float inv = __expf(-(float)j * 0.017988946039063842f); // 2*ln(1e4)/1024
    const int h = d >> 6, c = d & 63;

#pragma unroll
    for (int s4 = 0; s4 < 4; ++s4) {
#pragma unroll
        for (int r = 0; r < 8; ++r) {
            int tok = mbase + s4 * 16 + hi8 + r;     // token in [0, B*S)
            int b   = tok >> 11;
            int sp  = tok & (SEQ - 1);
            float ang = (float)sp * inv;
            float cs = __cosf(ang), sn = __sinf(ang);
            float v = acc[s4][r];
            float p = __shfl_xor(v, 1, 32);
            float res = (d & 1) ? fmaf(p, sn, v * cs)     // odd:  xe*sin + xo*cos
                                : fmaf(v, cs, -(p * sn)); // even: xe*cos - xo*sin
            if (!vtrans)
                dst[(((size_t)(b * NH + h) * SEQ + sp) * DK) + c] = (_Float16)res;
            else
                dst[(((size_t)(b * NH + h) * DK + c) * SEQ) + sp] = (_Float16)res;
        }
    }
}

// ---------------------------------------------------------------------------
// Attention helpers
// ---------------------------------------------------------------------------
__device__ __forceinline__ void attn_stage(
    const _Float16* __restrict__ ksrc0, const _Float16* __restrict__ ksrc1,
    const _Float16* __restrict__ vsrc0, const _Float16* __restrict__ vsrc1,
    _Float16* kd0, _Float16* kd1, _Float16* vd0, _Float16* vd1, int kv)
{
    v8h ka = *(const v8h*)(ksrc0 + (size_t)kv * DK);
    v8h kb = *(const v8h*)(ksrc1 + (size_t)kv * DK);
    v8h va = *(const v8h*)(vsrc0 + kv);
    v8h vb = *(const v8h*)(vsrc1 + kv);
    *(v8h*)kd0 = ka;
    *(v8h*)kd1 = kb;
    *(v8h*)vd0 = va;
    *(v8h*)vd1 = vb;
}

__device__ __forceinline__ void attn_step(
    const _Float16* Ksb, const _Float16* Vsb, _Float16* pw,
    const V16& aq0, const V16& aq1,
    v8f& O0, v8f& O1, v8f& O2, v8f& O3,
    float (&mr)[8], float (&lr)[8],
    int mrow, int o8, int o16, int hi8)
{
    // ---- scores: two 16x16 f32 tiles over 32 keys ----
    V16 b00, b01, b10, b11;
    b00.h[0] = *(const v8h*)(&Ksb[mrow * LDKC + o16]);
    b00.h[1] = *(const v8h*)(&Ksb[mrow * LDKC + o16 + 8]);
    b01.h[0] = *(const v8h*)(&Ksb[mrow * LDKC + 32 + o16]);
    b01.h[1] = *(const v8h*)(&Ksb[mrow * LDKC + 32 + o16 + 8]);
    b10.h[0] = *(const v8h*)(&Ksb[(16 + mrow) * LDKC + o16]);
    b10.h[1] = *(const v8h*)(&Ksb[(16 + mrow) * LDKC + o16 + 8]);
    b11.h[0] = *(const v8h*)(&Ksb[(16 + mrow) * LDKC + 32 + o16]);
    b11.h[1] = *(const v8h*)(&Ksb[(16 + mrow) * LDKC + 32 + o16 + 8]);

    v8f C0 = {}, C1 = {};
    C0 = wmma16(aq0.v, b00.v, C0);
    C0 = wmma16(aq1.v, b01.v, C0);
    C1 = wmma16(aq0.v, b10.v, C1);
    C1 = wmma16(aq1.v, b11.v, C1);
    // shape the score phase: batch the 8 DS reads, then the 4 WMMAs
    SCHED_GROUP(0x100, 8, 1);
    SCHED_GROUP(0x008, 4, 1);

    // ---- online softmax over the 32 columns ----
#pragma unroll
    for (int r = 0; r < 8; ++r) {
        float s0 = C0[r] * 0.125f;               // 1/sqrt(dk)
        float s1 = C1[r] * 0.125f;
        float t = fmaxf(s0, s1);
        t = fmaxf(t, __shfl_xor(t, 8, 32));
        t = fmaxf(t, __shfl_xor(t, 4, 32));
        t = fmaxf(t, __shfl_xor(t, 2, 32));
        t = fmaxf(t, __shfl_xor(t, 1, 32));
        float mn = fmaxf(mr[r], t);
        float al = __expf(mr[r] - mn);
        float p0 = __expf(s0 - mn);
        float p1 = __expf(s1 - mn);
        float rs = p0 + p1;
        rs += __shfl_xor(rs, 8, 32);
        rs += __shfl_xor(rs, 4, 32);
        rs += __shfl_xor(rs, 2, 32);
        rs += __shfl_xor(rs, 1, 32);
        lr[r] = lr[r] * al + rs;
        mr[r] = mn;
        O0[r] *= al; O1[r] *= al; O2[r] *= al; O3[r] *= al;
        // transpose P (C-layout -> A-layout) through per-wave LDS tile
        pw[(hi8 + r) * LDP + mrow]      = (_Float16)p0;
        pw[(hi8 + r) * LDP + 16 + mrow] = (_Float16)p1;
    }

    V16 ap;
    ap.h[0] = *(const v8h*)(pw + mrow * LDP + o16);
    ap.h[1] = *(const v8h*)(pw + mrow * LDP + o16 + 8);

    // ---- context accumulate: O += P @ V  (V^T rows are channels) ----
    V16 bv;
    bv.h[0] = *(const v8h*)(&Vsb[(0 * 16 + mrow) * LDVC + o16]);
    bv.h[1] = *(const v8h*)(&Vsb[(0 * 16 + mrow) * LDVC + o16 + 8]);
    O0 = wmma16(ap.v, bv.v, O0);
    bv.h[0] = *(const v8h*)(&Vsb[(1 * 16 + mrow) * LDVC + o16]);
    bv.h[1] = *(const v8h*)(&Vsb[(1 * 16 + mrow) * LDVC + o16 + 8]);
    O1 = wmma16(ap.v, bv.v, O1);
    bv.h[0] = *(const v8h*)(&Vsb[(2 * 16 + mrow) * LDVC + o16]);
    bv.h[1] = *(const v8h*)(&Vsb[(2 * 16 + mrow) * LDVC + o16 + 8]);
    O2 = wmma16(ap.v, bv.v, O2);
    bv.h[0] = *(const v8h*)(&Vsb[(3 * 16 + mrow) * LDVC + o16]);
    bv.h[1] = *(const v8h*)(&Vsb[(3 * 16 + mrow) * LDVC + o16 + 8]);
    O3 = wmma16(ap.v, bv.v, O3);
}

// ---------------------------------------------------------------------------
// Kernel 2: flash attention, double-buffered K/V LDS staging.
// grid: x = S/16/4 = 32, y = B*H = 32; 4 waves/block, wave = 16 query rows.
// Q,K: [B,H,S,dk] f16;  V: [B,H,dk,S] f16 (transposed);  CTX: [B,S,D] f16
// ---------------------------------------------------------------------------
__global__ __launch_bounds__(128) void attn(
    const _Float16* __restrict__ Q, const _Float16* __restrict__ K,
    const _Float16* __restrict__ V, _Float16* __restrict__ CTX)
{
    __shared__ __align__(16) _Float16 Ks0[32 * LDKC];
    __shared__ __align__(16) _Float16 Vs0[64 * LDVC];
    __shared__ __align__(16) _Float16 Ks1[32 * LDKC];
    __shared__ __align__(16) _Float16 Vs1[64 * LDVC];
    __shared__ __align__(16) _Float16 Pb[4][16 * LDP];

    const int tid  = threadIdx.x;
    const int w    = tid >> 5;
    const int lane = tid & 31;
    const int bh   = blockIdx.y;
    const int mbase = (blockIdx.x * 4 + w) * 16;

    const _Float16* qb = Q + (size_t)bh * SEQ * DK;
    const _Float16* kp = K + (size_t)bh * SEQ * DK;
    const _Float16* vp = V + (size_t)bh * DK * SEQ;

    const int mrow = lane & 15;
    const int o8   = (lane >> 4) * 8;
    const int o16  = (lane >> 4) * 16;
    const int hi8  = (lane >> 4) * 8;

    // staging source/dest offsets hoisted out of the kv loop
    const int i0 = tid, i1 = tid + 128;
    const _Float16* ksrc0 = kp + (size_t)(i0 >> 3) * DK + (i0 & 7) * 8;
    const _Float16* ksrc1 = kp + (size_t)(i1 >> 3) * DK + (i1 & 7) * 8;
    const _Float16* vsrc0 = vp + (size_t)(i0 >> 2) * SEQ + (i0 & 3) * 8;
    const _Float16* vsrc1 = vp + (size_t)(i1 >> 2) * SEQ + (i1 & 3) * 8;
    const int kdo0 = (i0 >> 3) * LDKC + (i0 & 7) * 8;
    const int kdo1 = (i1 >> 3) * LDKC + (i1 & 7) * 8;
    const int vdo0 = (i0 >> 2) * LDVC + (i0 & 3) * 8;
    const int vdo1 = (i1 >> 2) * LDVC + (i1 & 3) * 8;

    // Q A-fragments for d=[0,32) and [32,64): resident for the whole KV sweep
    V16 aq0, aq1;
    {
        const _Float16* qr = qb + (size_t)(mbase + mrow) * DK;
        aq0.h[0] = *(const v8h*)(qr + o16);
        aq0.h[1] = *(const v8h*)(qr + o16 + 8);
        aq1.h[0] = *(const v8h*)(qr + 32 + o16);
        aq1.h[1] = *(const v8h*)(qr + 32 + o16 + 8);
    }

    v8f O0 = {}, O1 = {}, O2 = {}, O3 = {};
    float mr[8], lr[8];
#pragma unroll
    for (int r = 0; r < 8; ++r) { mr[r] = -3.0e38f; lr[r] = 0.0f; }

    _Float16* pw = Pb[w];

    // prologue: stage chunk 0 into buffer 0
    attn_stage(ksrc0, ksrc1, vsrc0, vsrc1,
               &Ks0[kdo0], &Ks0[kdo1], &Vs0[vdo0], &Vs0[vdo1], 0);

    for (int kv = 0; kv < SEQ; kv += 64) {
        __syncthreads();   // buf0(kv) ready; all reads of buf1 finished
        if (kv + 32 < SEQ)
            attn_stage(ksrc0, ksrc1, vsrc0, vsrc1,
                       &Ks1[kdo0], &Ks1[kdo1], &Vs1[vdo0], &Vs1[vdo1], kv + 32);
        attn_step(Ks0, Vs0, pw, aq0, aq1, O0, O1, O2, O3, mr, lr,
                  mrow, o8, o16, hi8);

        __syncthreads();   // buf1(kv+32) ready; all reads of buf0 finished
        if (kv + 64 < SEQ)
            attn_stage(ksrc0, ksrc1, vsrc0, vsrc1,
                       &Ks0[kdo0], &Ks0[kdo1], &Vs0[vdo0], &Vs0[vdo1], kv + 64);
        attn_step(Ks1, Vs1, pw, aq0, aq1, O0, O1, O2, O3, mr, lr,
                  mrow, o8, o16, hi8);
    }

    // ---- normalize and store context in [B,S,D] f16 ----
    const int b = bh >> 4, h = bh & 15;
#pragma unroll
    for (int r = 0; r < 8; ++r) {
        float li = 1.0f / lr[r];
        int s = mbase + hi8 + r;
        _Float16* crow = CTX + ((size_t)(b * SEQ + s) * D_MODEL) + h * DK + mrow;
        crow[0]  = (_Float16)(O0[r] * li);
        crow[16] = (_Float16)(O1[r] * li);
        crow[32] = (_Float16)(O2[r] * li);
        crow[48] = (_Float16)(O3[r] * li);
    }
}

// ---------------------------------------------------------------------------
// Kernel 3: output projection OUT = CTX @ Wo^T (fp32 out), software-pipelined.
// grid: x = 16, y = B*S/64 = 64
// ---------------------------------------------------------------------------
__global__ __launch_bounds__(128) void out_proj(
    const _Float16* __restrict__ CTX, const _Float16* __restrict__ Wo,
    float* __restrict__ OUT)
{
    const int tid  = threadIdx.x;
    const int w    = tid >> 5;
    const int lane = tid & 31;
    const int nbase = (blockIdx.x * 4 + w) * 16;
    const int mbase = blockIdx.y * 64;
    const int mrow = lane & 15;
    const int o8   = (lane >> 4) * 8;
    const int o16  = (lane >> 4) * 16;
    const int hi8  = (lane >> 4) * 8;

    const _Float16* br  = Wo  + (size_t)(nbase + mrow) * D_MODEL;
    const _Float16* ar0 = CTX + (size_t)(mbase + mrow) * D_MODEL;

    v8f acc[4] = { {}, {}, {}, {} };

    V16 bc, ac[4];
    load_frags(ar0, br, 0, o8, o16, bc, ac);
    for (int kb = 32; kb < D_MODEL; kb += 32) {
        V16 bn, an[4];
        load_frags(ar0, br, kb, o8, o16, bn, an);
#pragma unroll
        for (int s = 0; s < 4; ++s)
            acc[s] = wmma16(ac[s].v, bc.v, acc[s]);
        SCHED_GROUP(0x020, 10, 0);
        SCHED_GROUP(0x008, 4, 0);
        bc = bn;
#pragma unroll
        for (int s = 0; s < 4; ++s) ac[s] = an[s];
    }
#pragma unroll
    for (int s = 0; s < 4; ++s)
        acc[s] = wmma16(ac[s].v, bc.v, acc[s]);

#pragma unroll
    for (int s = 0; s < 4; ++s) {
#pragma unroll
        for (int r = 0; r < 8; ++r)
            OUT[(size_t)(mbase + s * 16 + hi8 + r) * D_MODEL + nbase + mrow] = acc[s][r];
    }
}

// ---------------------------------------------------------------------------
extern "C" void kernel_launch(void* const* d_in, const int* in_sizes, int n_in,
                              void* d_out, int out_size, void* d_ws, size_t ws_size,
                              hipStream_t stream) {
    const float* Xq = (const float*)d_in[0];
    const float* Xk = (const float*)d_in[1];
    const float* Xv = (const float*)d_in[2];
    const float* Wq = (const float*)d_in[3];
    const float* Wk = (const float*)d_in[4];
    const float* Wv = (const float*)d_in[5];
    const float* Wo = (const float*)d_in[6];

    char* ws = (char*)d_ws;
    const size_t WB = (size_t)D_MODEL * D_MODEL * sizeof(_Float16);  // 2 MB
    const size_t TB = (size_t)NB * SEQ * D_MODEL * sizeof(_Float16); // 8 MB
    _Float16* wq16 = (_Float16*)(ws + 0 * WB);
    _Float16* wk16 = (_Float16*)(ws + 1 * WB);
    _Float16* wv16 = (_Float16*)(ws + 2 * WB);
    _Float16* wo16 = (_Float16*)(ws + 3 * WB);
    char* act = ws + 4 * WB;
    _Float16* xq16 = (_Float16*)(act + 0 * TB);
    _Float16* xk16 = (_Float16*)(act + 1 * TB);
    _Float16* xv16 = (_Float16*)(act + 2 * TB);
    _Float16* Qh   = (_Float16*)(act + 3 * TB);
    _Float16* Kh   = (_Float16*)(act + 4 * TB);
    _Float16* Vh   = (_Float16*)(act + 5 * TB);
    _Float16* Ctx  = (_Float16*)(act + 6 * TB);

    const int w4 = D_MODEL * D_MODEL / 4;        // float4 per weight
    const int x4 = NB * SEQ * D_MODEL / 4;       // float4 per activation tensor
    cvt16<<<w4 / 256, 256, 0, stream>>>((const float4*)Wq, (v4h*)wq16, w4);
    cvt16<<<w4 / 256, 256, 0, stream>>>((const float4*)Wk, (v4h*)wk16, w4);
    cvt16<<<w4 / 256, 256, 0, stream>>>((const float4*)Wv, (v4h*)wv16, w4);
    cvt16<<<w4 / 256, 256, 0, stream>>>((const float4*)Wo, (v4h*)wo16, w4);
    cvt16<<<x4 / 256, 256, 0, stream>>>((const float4*)Xq, (v4h*)xq16, x4);
    cvt16<<<x4 / 256, 256, 0, stream>>>((const float4*)Xk, (v4h*)xk16, x4);
    cvt16<<<x4 / 256, 256, 0, stream>>>((const float4*)Xv, (v4h*)xv16, x4);

    dim3 g1(D_MODEL / 64, NB * SEQ / 64, 3);   // (16, 64, 3)
    proj_rope<<<g1, 128, 0, stream>>>(xq16, xk16, xv16, wq16, wk16, wv16, Qh, Kh, Vh);

    dim3 g2(SEQ / 64, NB * NH);                // (32, 32)
    attn<<<g2, 128, 0, stream>>>(Qh, Kh, Vh, Ctx);

    dim3 g3(D_MODEL / 64, NB * SEQ / 64);      // (16, 64)
    out_proj<<<g3, 128, 0, stream>>>(Ctx, wo16, (float*)d_out);
}
